// GRUv10_15985868276401
// MI455X (gfx1250) — matrix operand order repaired and checked by reading, not compiled, not run on hardware
//
#include <hip/hip_runtime.h>

// ---------------------------------------------------------------------------
// Types for CDNA5 WMMA (bf16 -> f32, 16x16x32)
// ---------------------------------------------------------------------------
typedef __attribute__((ext_vector_type(16))) __bf16 v16bf;
typedef __attribute__((ext_vector_type(8)))  float  v8f;

union Frag {
    v16bf v;
    uint4 q[2];
};

__device__ __forceinline__ v8f wmma_bf16(v16bf a, v16bf b, v8f c) {
    return __builtin_amdgcn_wmma_f32_16x16x32_bf16(
        /*neg_a=*/false, a, /*neg_b=*/false, b,
        /*c_mod=*/(short)0, c, /*reuse_a=*/false, /*reuse_b=*/false);
}

// Problem constants
#define S_   256
#define B_   512
#define H_   512
#define D1_  64
#define SB_  131072   // S*B
#define G3_  1536     // 3*H

// ---------------------------------------------------------------------------
// Converters
// ---------------------------------------------------------------------------
__global__ void k_cvt_f32_bf16(const float* __restrict__ src,
                               __bf16* __restrict__ dst, long n) {
    long i = (long)blockIdx.x * blockDim.x + threadIdx.x;
    if (i < n) dst[i] = (__bf16)src[i];
}

// dst[r*cols+c] = src[r*stride + c]   (slice first `cols` of each row)
__global__ void k_slice_cvt(const float* __restrict__ src,
                            __bf16* __restrict__ dst,
                            long rows, int cols, int stride) {
    long i = (long)blockIdx.x * blockDim.x + threadIdx.x;
    if (i >= rows * (long)cols) return;
    long r = i / cols;
    int  c = (int)(i - r * cols);
    dst[i] = (__bf16)src[r * (long)stride + c];
}

// ---------------------------------------------------------------------------
// h0 = patient @ Wi2h^T + bi2h   (patient = X[0,:,64:70], B x 6; Wi2h: H x 6)
// ---------------------------------------------------------------------------
__global__ void k_h0(const float* __restrict__ X, const float* __restrict__ Wi2h,
                     const float* __restrict__ bi2h, float* __restrict__ h0) {
    int i = blockIdx.x * blockDim.x + threadIdx.x;
    if (i >= B_ * H_) return;
    int b = i >> 9, h = i & (H_ - 1);
    float acc = bi2h[h];
#pragma unroll
    for (int k = 0; k < 6; ++k)
        acc += X[(long)b * 70 + 64 + k] * Wi2h[h * 6 + k];
    h0[i] = acc;
}

__global__ void k_init_hx(const float* __restrict__ h0, float* __restrict__ hxf,
                          __bf16* __restrict__ hxb) {
    int i = blockIdx.x * blockDim.x + threadIdx.x;
    if (i >= B_ * H_) return;
    float v = h0[i];
    hxf[i] = v;
    hxb[i] = (__bf16)v;
}

// cfold[e] = bcomb2[e] + sum_k Wcomb2[e, H + k]   (ones-input fold)
__global__ void k_fold_comb2(const float* __restrict__ Wcomb2,
                             const float* __restrict__ bcomb2,
                             float* __restrict__ cfold) {
    int e = blockIdx.x * blockDim.x + threadIdx.x;
    if (e >= H_) return;
    float acc = bcomb2[e];
    for (int k = 0; k < H_; ++k) acc += Wcomb2[(long)e * 1024 + H_ + k];
    cfold[e] = acc;
}

// ---------------------------------------------------------------------------
// Generic WMMA GEMM:  C[M x N] (+)= A[M x K] @ W^T  (+ bias, + relu)
//   A  : bf16 row-major, leading dim lda
//   W  : bf16 row-major (N x K layout), leading dim ldw (supports col offset)
//   Cf : f32 out (optional), Cb : bf16 out (optional), leading dim ldc
// Block: 256 threads = 8 waves, tile 128(M) x 64(N); wave tile 16 x 64.
// Weight tile (64x32) staged through LDS (shared by all 8 waves).
// Requires: M % 128 == 0, N % 64 == 0, K % 32 == 0.
// ---------------------------------------------------------------------------
__global__ __launch_bounds__(256)
void k_gemm(const __bf16* __restrict__ A, int lda,
            const __bf16* __restrict__ W, int ldw,
            const float* __restrict__ bias,
            float* Cf, __bf16* Cb, int ldc,
            int K, int accumulate, int relu) {
    __shared__ __bf16 sB[64 * 32];

    const int tid  = threadIdx.x;
    const int lane = tid & 31;
    const int wid  = tid >> 5;
    const int m0 = blockIdx.y * 128 + wid * 16;
    const int n0 = blockIdx.x * 64;

    v8f acc[4] = {v8f{}, v8f{}, v8f{}, v8f{}};

    const int arow = m0 + (lane & 15);
    const int akb  = (lane >> 4) * 8;       // A K-offset per lane-half
    const int bkb  = (lane >> 4) * 8;       // B K-offset per lane-half
    const int fillIdx = tid * 8;            // 256 thr * 8 halves = 64x32 tile
    const int fN = fillIdx >> 5;
    const int fK = fillIdx & 31;

    for (int k = 0; k < K; k += 32) {
        __syncthreads();
        *(uint4*)&sB[fillIdx] =
            *(const uint4*)(W + (size_t)(n0 + fN) * ldw + k + fK);
        __syncthreads();

        Frag a;
        const __bf16* ap = A + (size_t)arow * lda + k + akb;
        a.q[0] = *(const uint4*)ap;
        a.q[1] = *(const uint4*)(ap + 16);

#pragma unroll
        for (int j = 0; j < 4; ++j) {
            Frag bf;
            const int nl = j * 16 + (lane & 15);
            bf.q[0] = *(const uint4*)&sB[nl * 32 + bkb];
            bf.q[1] = *(const uint4*)&sB[nl * 32 + bkb + 16];
            acc[j] = wmma_bf16(a.v, bf.v, acc[j]);
        }
    }

#pragma unroll
    for (int j = 0; j < 4; ++j) {
        const int col = n0 + j * 16 + (lane & 15);
        const float bv = bias ? bias[col] : 0.0f;
#pragma unroll
        for (int v = 0; v < 8; ++v) {
            const int row = m0 + v + ((lane >> 4) << 3);
            float x = acc[j][v] + bv;
            if (accumulate) x += Cf[(size_t)row * ldc + col];
            if (relu) x = fmaxf(x, 0.0f);
            if (Cf) Cf[(size_t)row * ldc + col] = x;
            if (Cb) Cb[(size_t)row * ldc + col] = (__bf16)x;
        }
    }
}

// ---------------------------------------------------------------------------
// One GRU time step, fused:  gh = hx @ Whh^T + bhh; gates; masked update.
// Block: 256 threads = 8 waves; output tile 32(B) x 64(H); wave: 16x16 x 3 gates.
// Grid: (H/64, B/32). hx ping-pongs (src -> dst) to avoid cross-block hazards.
// ---------------------------------------------------------------------------
__global__ __launch_bounds__(256)
void k_gru_step(const float* __restrict__ gi_t,       // B x 3H (f32)
                const float* __restrict__ hx_src_f,   // B x H
                const __bf16* __restrict__ hx_src_b,  // B x H
                float* __restrict__ hx_dst_f,
                __bf16* __restrict__ hx_dst_b,
                const __bf16* __restrict__ Whh,       // 3H x H (bf16)
                const float* __restrict__ bhh,        // 3H
                const float* __restrict__ mask_t,     // B
                __bf16* __restrict__ hs_t)            // B x H
{
    const int tid  = threadIdx.x;
    const int lane = tid & 31;
    const int wid  = tid >> 5;
    const int bb0 = blockIdx.y * 32 + (wid & 1) * 16;
    const int hh0 = blockIdx.x * 64 + (wid >> 1) * 16;

    v8f aR = v8f{}, aZ = v8f{}, aN = v8f{};

    const int arow = bb0 + (lane & 15);
    const int akb  = (lane >> 4) * 8;
    const int ncol = hh0 + (lane & 15);
    const int bkb  = (lane >> 4) * 8;

    for (int k = 0; k < H_; k += 32) {
        Frag a;
        const __bf16* ap = hx_src_b + (size_t)arow * H_ + k + akb;
        a.q[0] = *(const uint4*)ap;
        a.q[1] = *(const uint4*)(ap + 16);

        Frag fr, fz, fn;
        const __bf16* p = Whh + (size_t)ncol * H_ + k + bkb;
        fr.q[0] = *(const uint4*)p;
        fr.q[1] = *(const uint4*)(p + 16);
        p += (size_t)H_ * H_;
        fz.q[0] = *(const uint4*)p;
        fz.q[1] = *(const uint4*)(p + 16);
        p += (size_t)H_ * H_;
        fn.q[0] = *(const uint4*)p;
        fn.q[1] = *(const uint4*)(p + 16);

        aR = wmma_bf16(a.v, fr.v, aR);
        aZ = wmma_bf16(a.v, fz.v, aZ);
        aN = wmma_bf16(a.v, fn.v, aN);
    }

    const int hh = hh0 + (lane & 15);
    const float bhr = bhh[hh], bhz = bhh[H_ + hh], bhn = bhh[2 * H_ + hh];

#pragma unroll
    for (int v = 0; v < 8; ++v) {
        const int b = bb0 + v + ((lane >> 4) << 3);
        const float ir  = gi_t[(size_t)b * G3_ + hh];
        const float iz  = gi_t[(size_t)b * G3_ + H_ + hh];
        const float inn = gi_t[(size_t)b * G3_ + 2 * H_ + hh];
        const float r = 1.0f / (1.0f + expf(-(ir + aR[v] + bhr)));
        const float z = 1.0f / (1.0f + expf(-(iz + aZ[v] + bhz)));
        const float n = tanhf(inn + r * (aN[v] + bhn));
        const float hp = hx_src_f[(size_t)b * H_ + hh];
        const float hn = (1.0f - z) * n + z * hp;
        const float m  = mask_t[b];
        const float ho = hn * m + hp * (1.0f - m);
        hx_dst_f[(size_t)b * H_ + hh] = ho;
        const __bf16 hb = (__bf16)ho;
        hx_dst_b[(size_t)b * H_ + hh] = hb;
        hs_t[(size_t)b * H_ + hh]     = hb;
    }
}

// ---------------------------------------------------------------------------
// LayerNorm over last dim = 512; one 256-thread block per row; bf16 output.
// ---------------------------------------------------------------------------
__global__ __launch_bounds__(256)
void k_ln512(const float* __restrict__ src, const float* __restrict__ g,
             const float* __restrict__ b, __bf16* __restrict__ dst) {
    __shared__ float s[256];
    const int row = blockIdx.x;
    const int tid = threadIdx.x;
    const float* x = src + (size_t)row * H_;
    const float a0 = x[tid], a1 = x[tid + 256];

    s[tid] = a0 + a1;
    __syncthreads();
    for (int o = 128; o > 0; o >>= 1) {
        if (tid < o) s[tid] += s[tid + o];
        __syncthreads();
    }
    const float mean = s[0] * (1.0f / 512.0f);
    __syncthreads();

    const float d0 = a0 - mean, d1 = a1 - mean;
    s[tid] = d0 * d0 + d1 * d1;
    __syncthreads();
    for (int o = 128; o > 0; o >>= 1) {
        if (tid < o) s[tid] += s[tid + o];
        __syncthreads();
    }
    const float inv = rsqrtf(s[0] * (1.0f / 512.0f) + 1e-5f);

    dst[(size_t)row * H_ + tid]       = (__bf16)(d0 * inv * g[tid] + b[tid]);
    dst[(size_t)row * H_ + tid + 256] = (__bf16)(d1 * inv * g[tid + 256] + b[tid + 256]);
}

// ---------------------------------------------------------------------------
// Head: y = relu(x @ Wnn1^T + bnn1); LN8(g3,b3); out = y @ Wnn2^T + bnn2.
// One wave per (s,b) row; 8 waves per block.
// ---------------------------------------------------------------------------
__global__ __launch_bounds__(256)
void k_head(const __bf16* __restrict__ X, const float* __restrict__ Wnn1,
            const float* __restrict__ bnn1, const float* __restrict__ g3,
            const float* __restrict__ b3, const float* __restrict__ Wnn2,
            const float* __restrict__ bnn2, float* __restrict__ out) {
    const int lane = threadIdx.x & 31;
    const int row  = blockIdx.x * 8 + (threadIdx.x >> 5);
    const __bf16* x = X + (size_t)row * H_;

    float acc[8] = {0, 0, 0, 0, 0, 0, 0, 0};
    for (int k = lane; k < H_; k += 32) {
        const float xv = (float)x[k];
#pragma unroll
        for (int o = 0; o < 8; ++o) acc[o] += xv * Wnn1[o * H_ + k];
    }
#pragma unroll
    for (int o = 0; o < 8; ++o)
        for (int off = 16; off > 0; off >>= 1)
            acc[o] += __shfl_xor(acc[o], off, 32);

    if (lane == 0) {
        float y[8], mean = 0.0f;
#pragma unroll
        for (int o = 0; o < 8; ++o) {
            y[o] = fmaxf(acc[o] + bnn1[o], 0.0f);
            mean += y[o];
        }
        mean *= 0.125f;
        float var = 0.0f;
#pragma unroll
        for (int o = 0; o < 8; ++o) {
            const float d = y[o] - mean;
            var += d * d;
        }
        const float inv = rsqrtf(var * 0.125f + 1e-5f);
        float o0 = bnn2[0], o1 = bnn2[1];
#pragma unroll
        for (int o = 0; o < 8; ++o) {
            const float z = (y[o] - mean) * inv * g3[o] + b3[o];
            o0 += z * Wnn2[o];
            o1 += z * Wnn2[8 + o];
        }
        out[(size_t)row * 2]     = o0;
        out[(size_t)row * 2 + 1] = o1;
    }
}

// ---------------------------------------------------------------------------
// Host orchestration
// ---------------------------------------------------------------------------
extern "C" void kernel_launch(void* const* d_in, const int* in_sizes, int n_in,
                              void* d_out, int out_size, void* d_ws, size_t ws_size,
                              hipStream_t stream) {
    (void)in_sizes; (void)n_in; (void)out_size; (void)ws_size;

    const float* X      = (const float*)d_in[0];
    const float* Mm     = (const float*)d_in[1];
    const float* mask   = (const float*)d_in[2];
    const float* Wi2h   = (const float*)d_in[3];
    const float* bi2h   = (const float*)d_in[4];
    const float* Wcomb1 = (const float*)d_in[5];
    const float* bcomb1 = (const float*)d_in[6];
    const float* Wih1   = (const float*)d_in[7];
    const float* Whh1   = (const float*)d_in[8];
    const float* bih1   = (const float*)d_in[9];
    const float* bhh1   = (const float*)d_in[10];
    const float* Wc21   = (const float*)d_in[11];
    const float* bc21   = (const float*)d_in[12];
    const float* Wcomb2 = (const float*)d_in[13];
    const float* bcomb2 = (const float*)d_in[14];
    const float* Wih2   = (const float*)d_in[15];
    const float* Whh2   = (const float*)d_in[16];
    const float* bih2   = (const float*)d_in[17];
    const float* bhh2   = (const float*)d_in[18];
    const float* Wc22   = (const float*)d_in[19];
    const float* bc22   = (const float*)d_in[20];
    const float* g1     = (const float*)d_in[21];
    const float* b1     = (const float*)d_in[22];
    const float* g2     = (const float*)d_in[23];
    const float* b2     = (const float*)d_in[24];
    const float* g3     = (const float*)d_in[25];
    const float* b3     = (const float*)d_in[26];
    const float* Wnn1   = (const float*)d_in[27];
    const float* bnn1   = (const float*)d_in[28];
    const float* Wnn2   = (const float*)d_in[29];
    const float* bnn2   = (const float*)d_in[30];
    float* out = (float*)d_out;

    // Workspace carve-up
    char* w = (char*)d_ws;
    auto alloc = [&](size_t bytes) -> char* {
        char* p = w;
        w += (bytes + 255) & ~(size_t)255;
        return p;
    };
    const size_t SB = SB_;
    __bf16* xin_bf  = (__bf16*)alloc(SB * D1_ * 2);
    __bf16* min_bf  = (__bf16*)alloc(SB * D1_ * 2);
    __bf16* comb_bf = (__bf16*)alloc(SB * H_ * 2);
    float*  fbuf    = (float*) alloc(SB * H_ * 4);      // f32 scratch (comb1 / out)
    float*  gi      = (float*) alloc(SB * (size_t)G3_ * 4);
    __bf16* hs_bf   = (__bf16*)alloc(SB * H_ * 2);
    __bf16* ln_bf   = (__bf16*)alloc(SB * H_ * 2);
    float*  hxf0    = (float*) alloc(B_ * H_ * 4);
    float*  hxf1    = (float*) alloc(B_ * H_ * 4);
    __bf16* hxb0    = (__bf16*)alloc(B_ * H_ * 2);
    __bf16* hxb1    = (__bf16*)alloc(B_ * H_ * 2);
    float*  h0f     = (float*) alloc(B_ * H_ * 4);
    __bf16* wcomb1b = (__bf16*)alloc((size_t)D1_ * 128 * 2);
    __bf16* wih1b   = (__bf16*)alloc((size_t)G3_ * D1_ * 2);
    __bf16* whh1b   = (__bf16*)alloc((size_t)G3_ * H_ * 2);
    __bf16* wc21b   = (__bf16*)alloc((size_t)H_ * 576 * 2);
    __bf16* wcomb2b = (__bf16*)alloc((size_t)H_ * 1024 * 2);
    __bf16* wih2b   = (__bf16*)alloc((size_t)G3_ * H_ * 2);
    __bf16* whh2b   = (__bf16*)alloc((size_t)G3_ * H_ * 2);
    __bf16* wc22b   = (__bf16*)alloc((size_t)H_ * 1024 * 2);
    float*  cfold   = (float*) alloc(H_ * 4);

    float*  hxf[2] = {hxf0, hxf1};
    __bf16* hxb[2] = {hxb0, hxb1};

    auto cvt = [&](const float* s, __bf16* d, long n) {
        k_cvt_f32_bf16<<<(unsigned)((n + 255) / 256), 256, 0, stream>>>(s, d, n);
    };

    // --- input + weight conversion ---
    {
        long n = (long)SB * D1_;
        k_slice_cvt<<<(unsigned)((n + 255) / 256), 256, 0, stream>>>(X,  xin_bf, SB, D1_, 70);
        k_slice_cvt<<<(unsigned)((n + 255) / 256), 256, 0, stream>>>(Mm, min_bf, SB, D1_, 70);
    }
    cvt(Wcomb1, wcomb1b, (long)D1_ * 128);
    cvt(Wih1,   wih1b,   (long)G3_ * D1_);
    cvt(Whh1,   whh1b,   (long)G3_ * H_);
    cvt(Wc21,   wc21b,   (long)H_ * 576);
    cvt(Wcomb2, wcomb2b, (long)H_ * 1024);
    cvt(Wih2,   wih2b,   (long)G3_ * H_);
    cvt(Whh2,   whh2b,   (long)G3_ * H_);
    cvt(Wc22,   wc22b,   (long)H_ * 1024);

    // --- h0 ---
    k_h0<<<(B_ * H_) / 256, 256, 0, stream>>>(X, Wi2h, bi2h, h0f);

    const dim3 blk(256);
    auto gemm = [&](const __bf16* A, int lda, const __bf16* W, int ldw,
                    const float* bias, float* Cf, __bf16* Cb, int ldc,
                    int N, int K, int acc, int relu) {
        dim3 grid(N / 64, SB_ / 128);
        k_gemm<<<grid, blk, 0, stream>>>(A, lda, W, ldw, bias, Cf, Cb, ldc, K, acc, relu);
    };

    // ============== Layer 1 ==============
    // comb1 = Xin @ Wcomb1[:, :64]^T + bcomb1 + Min @ Wcomb1[:, 64:]^T
    gemm(xin_bf, D1_, wcomb1b,      128, bcomb1, fbuf, nullptr, D1_, D1_, D1_, 0, 0);
    gemm(min_bf, D1_, wcomb1b + 64, 128, nullptr, fbuf, comb_bf, D1_, D1_, D1_, 1, 0);
    // gi1 = comb1 @ Wih1^T + bih1   (f32, SB x 1536)
    gemm(comb_bf, D1_, wih1b, D1_, bih1, gi, nullptr, G3_, G3_, D1_, 0, 0);

    // scan layer 1
    k_init_hx<<<(B_ * H_) / 256, 256, 0, stream>>>(h0f, hxf[0], hxb[0]);
    for (int t = 0; t < S_; ++t) {
        const int s = t & 1, d = 1 - s;
        k_gru_step<<<dim3(H_ / 64, B_ / 32), blk, 0, stream>>>(
            gi + (size_t)t * B_ * G3_, hxf[s], hxb[s], hxf[d], hxb[d],
            whh1b, bhh1, mask + (size_t)t * B_, hs_bf + (size_t)t * B_ * H_);
    }

    // out1 = comb1 @ Wc21[:, :64]^T + bc21 + hs1 @ Wc21[:, 64:]^T
    gemm(comb_bf, D1_, wc21b,      576, bc21,   fbuf, nullptr, H_, H_, D1_, 0, 0);
    gemm(hs_bf,   H_,  wc21b + 64, 576, nullptr, fbuf, nullptr, H_, H_, H_,  1, 0);
    // LN1 -> ln_bf
    k_ln512<<<SB_, 256, 0, stream>>>(fbuf, g1, b1, ln_bf);

    // ============== Layer 2 ==============
    // ones-input fold: cfold = bcomb2 + rowsum(Wcomb2[:, H:])
    k_fold_comb2<<<H_ / 256, 256, 0, stream>>>(Wcomb2, bcomb2, cfold);
    // comb2 = ln1 @ Wcomb2[:, :H]^T + cfold   (bf16 only)
    gemm(ln_bf, H_, wcomb2b, 1024, cfold, nullptr, comb_bf, H_, H_, H_, 0, 0);
    // gi2 = comb2 @ Wih2^T + bih2
    gemm(comb_bf, H_, wih2b, H_, bih2, gi, nullptr, G3_, G3_, H_, 0, 0);

    // scan layer 2
    k_init_hx<<<(B_ * H_) / 256, 256, 0, stream>>>(h0f, hxf[0], hxb[0]);
    for (int t = 0; t < S_; ++t) {
        const int s = t & 1, d = 1 - s;
        k_gru_step<<<dim3(H_ / 64, B_ / 32), blk, 0, stream>>>(
            gi + (size_t)t * B_ * G3_, hxf[s], hxb[s], hxf[d], hxb[d],
            whh2b, bhh2, mask + (size_t)t * B_, hs_bf + (size_t)t * B_ * H_);
    }

    // out2 = comb2 @ Wc22[:, :H]^T + bc22 + hs2 @ Wc22[:, H:]^T
    gemm(comb_bf, H_, wc22b,       1024, bc22,   fbuf, nullptr, H_, H_, H_, 0, 0);
    gemm(hs_bf,   H_, wc22b + 512, 1024, nullptr, fbuf, nullptr, H_, H_, H_, 1, 0);
    // LN2 -> ln_bf
    k_ln512<<<SB_, 256, 0, stream>>>(fbuf, g2, b2, ln_bf);

    // Head: relu(@Wnn1^T)+LN8+@Wnn2^T -> out (S,B,2)
    k_head<<<SB_ / 8, 256, 0, stream>>>(ln_bf, Wnn1, bnn1, g3, b3, Wnn2, bnn2, out);
}
